// EnhancedLSTM_56616258896274
// MI455X (gfx1250) — compile-verified, matrix operand order checked by reading, not run on hardware
//
#include <hip/hip_runtime.h>
#include <hip/hip_bf16.h>

#define BB 128
#define TT 512
#define DD 128
#define HH 512
#define FH 2048           // 4*H
#define MM (BB*TT)        // 65536 rows for time-parallel GEMMs
#define EPSF 1e-3f
#define BK 128            // K-chunk staged in LDS
#define LSTR (BK + 8)     // padded LDS row stride (elements): 272B -> conflict-free b128 reads

typedef __attribute__((ext_vector_type(16))) __bf16        v16bf;
typedef __attribute__((ext_vector_type(8)))  float         v8f;
typedef __attribute__((ext_vector_type(4)))  unsigned int  uivec4;
typedef __attribute__((ext_vector_type(4)))  int           i4v;

union Frag { v16bf v; uivec4 u[2]; };

#if defined(__has_builtin)
#  if __has_builtin(__builtin_amdgcn_global_load_async_to_lds_b128)
#    define ASYNC_LDS 1
#  endif
#  if __has_builtin(__builtin_amdgcn_s_wait_asynccnt)
#    define HAVE_WAIT_ASYNC 1
#  endif
#endif

typedef __attribute__((address_space(1))) i4v* gptr_t;   // global int4*
typedef __attribute__((address_space(3))) i4v* lptr_t;   // LDS int4*

// 16-byte global -> LDS copy; async (ASYNCcnt) when the toolchain exposes it.
__device__ __forceinline__ void copy16_to_lds(unsigned short* l, const unsigned short* g) {
#ifdef ASYNC_LDS
  __builtin_amdgcn_global_load_async_to_lds_b128(
      (gptr_t)(unsigned long long)g,
      (lptr_t)(unsigned int)(unsigned long long)l, 0, 0);
#else
  *(uivec4*)l = *(const uivec4*)g;      // global_load_b128 + ds_store_b128
#endif
}
__device__ __forceinline__ void wait_async_copies() {
#ifdef ASYNC_LDS
#  ifdef HAVE_WAIT_ASYNC
  __builtin_amdgcn_s_wait_asynccnt(0);
#  else
  asm volatile("s_wait_asynccnt 0x0" ::: "memory");
#  endif
#endif
}

__device__ __forceinline__ unsigned short f2bf(float f) {
  unsigned int u = __float_as_uint(f);
  u += 0x7FFFu + ((u >> 16) & 1u);          // round-to-nearest-even
  return (unsigned short)(u >> 16);
}
__device__ __forceinline__ float sigf(float x)   { return 1.f / (1.f + __expf(-x)); }
__device__ __forceinline__ float tanh_f(float x) { return 2.f / (1.f + __expf(-2.f * x)) - 1.f; }

__device__ __forceinline__ v8f wmma_bf16(Frag a, Frag b, v8f c) {
  // D = A(16x32 bf16) * B(32x16 bf16) + C(16x16 f32)
  return __builtin_amdgcn_wmma_f32_16x16x32_bf16(false, a.v, false, b.v, (short)0, c, false, false);
}

// ---------------------------------------------------------------------------
// Utility kernels
// ---------------------------------------------------------------------------
__global__ void fill_u32(unsigned int* p, unsigned int v, unsigned int n) {
  unsigned int i = blockIdx.x * blockDim.x + threadIdx.x;
  if (i < n) p[i] = v;
}

__global__ void cast_bf16(unsigned short* dst, const float* src, unsigned int n) {
  unsigned int i = blockIdx.x * blockDim.x + threadIdx.x;
  if (i < n) dst[i] = f2bf(src[i]);
}

// dst[N][K] (bf16) = transpose of src[K][N] (f32)
__global__ void transpose_bf16(unsigned short* dst, const float* src, int K, int N) {
  unsigned int idx = blockIdx.x * blockDim.x + threadIdx.x;
  if (idx >= (unsigned int)(K * N)) return;
  int nn = idx / K, kk = idx % K;
  dst[idx] = f2bf(src[(size_t)kk * N + nn]);
}

// ---------------------------------------------------------------------------
// Tiled WMMA GEMM: C[M,N](f32) = A[M,K](bf16 row-major) @ Bt[N,K](bf16) + bias
// Block = 8 waves covering 128(M) x 64(N).  grid = (N/64, M/128).
// B strip staged in LDS (async copies, double-buffered K-chunks of 128).
// Inner loop is a 2-stage register pipeline: fragments for K-step k+32 are
// loaded before the WMMAs of K-step k, so ds/global latency overlaps WMMA.
// ---------------------------------------------------------------------------
__global__ void gemm_bf16(float* __restrict__ C,
                          const unsigned short* __restrict__ A,
                          const unsigned short* __restrict__ Bt,
                          const float* __restrict__ bias,
                          int M, int N, int K) {
  __shared__ __attribute__((aligned(16))) unsigned short sb[2][64 * LSTR];
  const int tid   = threadIdx.x;
  const int lane  = tid & 31;
  const int wave  = tid >> 5;
  const int mbase = blockIdx.y * 128 + wave * 16;
  const int nbase = blockIdx.x * 64;
  const int mrow  = mbase + (lane & 15);
  const int khalf = (lane >> 4) * 8;          // lanes 0-15: K 0-7/16-23; 16-31: K 8-15/24-31

  // cooperative stage of one 64 x BK chunk of Bt into sb[buf]
  auto stage = [&](int buf, int kc) {
#pragma unroll
    for (int i = 0; i < 4; ++i) {
      const int c = tid + i * 256;            // 0..1023: 16B segments
      const int row = c >> 4, seg = c & 15;
      copy16_to_lds(&sb[buf][row * LSTR + seg * 8],
                    Bt + (size_t)(nbase + row) * K + kc + seg * 8);
    }
  };

  v8f acc[4] = {};
  const int nchunks = K / BK;
  stage(0, 0);
  wait_async_copies();
  __syncthreads();

  for (int ch = 0; ch < nchunks; ++ch) {
    const int kc = ch * BK;
    if (ch + 1 < nchunks) {
      stage((ch + 1) & 1, kc + BK);
      __builtin_prefetch(A + (size_t)mrow * K + kc + BK, 0, 1);
    }
    const unsigned short* sbuf = sb[ch & 1];

    Frag aF[2]; Frag bF[2][4];
    auto loadfr = [&](int slot, int k) {
      const unsigned short* ap = A + (size_t)mrow * K + kc + k + khalf;
      aF[slot].u[0] = *(const uivec4*)ap;
      aF[slot].u[1] = *(const uivec4*)(ap + 16);
#pragma unroll
      for (int j = 0; j < 4; ++j) {
        const unsigned short* lrow = sbuf + (j * 16 + (lane & 15)) * LSTR + k + khalf;
        bF[slot][j].u[0] = *(const uivec4*)lrow;
        bF[slot][j].u[1] = *(const uivec4*)(lrow + 16);
      }
    };

    loadfr(0, 0);
#pragma unroll
    for (int k = 0; k < BK; k += 32) {
      const int cur = (k >> 5) & 1;
      if (k + 32 < BK) loadfr(cur ^ 1, k + 32);   // next-step loads before this step's WMMAs
#pragma unroll
      for (int j = 0; j < 4; ++j)
        acc[j] = wmma_bf16(aF[cur], bF[cur][j], acc[j]);
    }
    wait_async_copies();
    __syncthreads();
  }

  const int rbase = mbase + (lane >> 4) * 8;
#pragma unroll
  for (int j = 0; j < 4; ++j) {
    const int ncol = nbase + j * 16 + (lane & 15);
    const float bv = bias ? bias[ncol] : 0.f;
#pragma unroll
    for (int r = 0; r < 8; ++r)
      C[(size_t)(rbase + r) * N + ncol] = acc[j][r] + bv;
  }
}

// ---------------------------------------------------------------------------
// Fused LSTM step:  z = xz[:,t,:] + h_{t-1} @ U ; gates ; c,h update ; BN epilogue
// Wave computes 16 batch rows x 16 hidden cols for all 4 gates.  Block = 8
// waves (all 128 batch rows, same 16 hidden cols); the 64-row U strip is
// staged in LDS (double-buffered chunks).  2-stage register pipeline in the
// K-loop.  grid = H/16 = 32 blocks.  h state double-buffered (hr -> hw).
// ---------------------------------------------------------------------------
__global__ void lstm_step(const float* __restrict__ xz,           // [B*T, 4H] f32
                          const unsigned short* __restrict__ Ut,  // [4H, H] bf16 (transposed U)
                          const unsigned short* __restrict__ hr,  // [B, H] bf16 read state
                          unsigned short* __restrict__ hw,        // [B, H] bf16 write state
                          float* __restrict__ cstate,             // [B, H] f32
                          unsigned short* __restrict__ hseq_bf,   // [B*T, H] bf16
                          float* __restrict__ hseq_f32,           // [B*T, H] f32 or nullptr
                          const float* __restrict__ bn_g, const float* __restrict__ bn_b,
                          const float* __restrict__ bn_m, const float* __restrict__ bn_v,
                          int t) {
  __shared__ __attribute__((aligned(16))) unsigned short sb[2][64 * LSTR];
  const int tid   = threadIdx.x;
  const int lane  = tid & 31;
  const int wave  = tid >> 5;
  const int mbase = wave * 16;                       // batch tile
  const int nbase = blockIdx.x * 16;
  const int n     = nbase + (lane & 15);             // hidden unit
  const int khalf = (lane >> 4) * 8;
  const int mrow  = mbase + (lane & 15);
  const int rbase = mbase + (lane >> 4) * 8;

  // stage one (4*16 rows) x BK chunk of the per-block U strip
  auto stage = [&](int buf, int kc) {
#pragma unroll
    for (int i = 0; i < 4; ++i) {
      const int c = tid + i * 256;                   // 0..1023
      const int row = c >> 4, seg = c & 15;          // row = g*16 + rr
      const int g = row >> 4, rr = row & 15;
      copy16_to_lds(&sb[buf][row * LSTR + seg * 8],
                    Ut + (size_t)(g * HH + nbase + rr) * HH + kc + seg * 8);
    }
  };

  // init accumulators from precomputed input projection xz (C-operand layout)
  v8f acc[4];
#pragma unroll
  for (int g = 0; g < 4; ++g)
#pragma unroll
    for (int r = 0; r < 8; ++r)
      acc[g][r] = xz[((size_t)(rbase + r) * TT + t) * FH + g * HH + n];

  stage(0, 0);
  wait_async_copies();
  __syncthreads();

  for (int ch = 0; ch < HH / BK; ++ch) {
    const int kc = ch * BK;
    if (ch + 1 < HH / BK) stage((ch + 1) & 1, kc + BK);
    const unsigned short* sbuf = sb[ch & 1];

    Frag aF[2]; Frag bF[2][4];
    auto loadfr = [&](int slot, int k) {
      const unsigned short* ap = hr + (size_t)mrow * HH + kc + k + khalf;
      aF[slot].u[0] = *(const uivec4*)ap;
      aF[slot].u[1] = *(const uivec4*)(ap + 16);
#pragma unroll
      for (int g = 0; g < 4; ++g) {
        const unsigned short* lrow = sbuf + (g * 16 + (lane & 15)) * LSTR + k + khalf;
        bF[slot][g].u[0] = *(const uivec4*)lrow;
        bF[slot][g].u[1] = *(const uivec4*)(lrow + 16);
      }
    };

    loadfr(0, 0);
#pragma unroll
    for (int k = 0; k < BK; k += 32) {
      const int cur = (k >> 5) & 1;
      if (k + 32 < BK) loadfr(cur ^ 1, k + 32);
#pragma unroll
      for (int g = 0; g < 4; ++g)
        acc[g] = wmma_bf16(aF[cur], bF[cur][g], acc[g]);
    }
    wait_async_copies();
    __syncthreads();
  }

  float scale = 1.f, shift = 0.f;
  if (bn_g) {
    scale = bn_g[n] * rsqrtf(bn_v[n] + EPSF);
    shift = bn_b[n] - bn_m[n] * scale;
  }
#pragma unroll
  for (int r = 0; r < 8; ++r) {
    const int m = rbase + r;
    const float iv = sigf(acc[0][r]);
    const float fv = sigf(acc[1][r]);
    const float gv = tanh_f(acc[2][r]);
    const float ov = sigf(acc[3][r]);
    const size_t sidx = (size_t)m * HH + n;
    const float c = fv * cstate[sidx] + iv * gv;
    const float h = ov * tanh_f(c);
    cstate[sidx] = c;
    hw[sidx]     = f2bf(h);                    // next-step A operand
    const size_t qidx = ((size_t)m * TT + t) * HH + n;
    hseq_bf[qidx] = f2bf(bn_g ? (h * scale + shift) : h);
    if (hseq_f32) hseq_f32[qidx] = h;
  }
}

// ---------------------------------------------------------------------------
// Attention: s[row] = sum_k tanh(e[row,k])   (one wave per row, wave32 reduce)
// ---------------------------------------------------------------------------
__global__ void attn_score(float* __restrict__ s, const float* __restrict__ e) {
  const int row  = blockIdx.x * 8 + (threadIdx.x >> 5);
  const int lane = threadIdx.x & 31;
  const float* er = e + (size_t)row * HH;
  float acc = 0.f;
  for (int c = lane; c < HH; c += 32) acc += tanh_f(er[c]);
#pragma unroll
  for (int off = 16; off; off >>= 1) acc += __shfl_xor(acc, off, 32);
  if (lane == 0) s[row] = acc;
}

// softmax over time (T=512) per batch row; one block (128 threads) per batch
__global__ void softmax_t(float* __restrict__ a, const float* __restrict__ s) {
  __shared__ float red[128];
  const int b = blockIdx.x, tid = threadIdx.x;
  const float* sb = s + (size_t)b * TT;
  float mx = -1e30f;
  for (int t = tid; t < TT; t += 128) mx = fmaxf(mx, sb[t]);
  red[tid] = mx; __syncthreads();
  for (int w = 64; w; w >>= 1) { if (tid < w) red[tid] = fmaxf(red[tid], red[tid + w]); __syncthreads(); }
  mx = red[0]; __syncthreads();
  float sum = 0.f;
  for (int t = tid; t < TT; t += 128) sum += __expf(sb[t] - mx);
  red[tid] = sum; __syncthreads();
  for (int w = 64; w; w >>= 1) { if (tid < w) red[tid] += red[tid + w]; __syncthreads(); }
  const float inv = 1.f / red[0];
  for (int t = tid; t < TT; t += 128) a[(size_t)b * TT + t] = __expf(sb[t] - mx) * inv;
}

// pooled[b,h] = (1/T) * sum_t a[b,t] * hseq[b,t,h]
__global__ void weighted_mean(float* __restrict__ pooled,
                              const float* __restrict__ hseq,
                              const float* __restrict__ a) {
  const int idx = blockIdx.x * blockDim.x + threadIdx.x;   // b*H + hc
  const int b = idx >> 9, hc = idx & 511;
  const float* hb = hseq + (size_t)b * TT * HH + hc;
  const float* ab = a + (size_t)b * TT;
  float acc = 0.f;
  for (int t = 0; t < TT; ++t) acc += hb[(size_t)t * HH] * ab[t];
  pooled[idx] = acc * (1.f / (float)TT);
}

// small dense: out[M,N] = act(in[M,K] @ W[K,N] + bias), optional relu + BN
__global__ void dense_k(float* __restrict__ out, const float* __restrict__ in,
                        const float* __restrict__ W, const float* __restrict__ bias,
                        int M, int K, int N, int relu,
                        const float* bng, const float* bnb, const float* bnm, const float* bnv) {
  const int idx = blockIdx.x * blockDim.x + threadIdx.x;
  if (idx >= M * N) return;
  const int m = idx / N, n = idx % N;
  float acc = bias[n];
  for (int k = 0; k < K; ++k) acc += in[(size_t)m * K + k] * W[(size_t)k * N + n];
  if (relu) acc = fmaxf(acc, 0.f);
  if (bng)  acc = (acc - bnm[n]) * rsqrtf(bnv[n] + EPSF) * bng[n] + bnb[n];
  out[idx] = acc;
}

// ---------------------------------------------------------------------------
// Host orchestration
// ---------------------------------------------------------------------------
extern "C" void kernel_launch(void* const* d_in, const int* in_sizes, int n_in,
                              void* d_out, int out_size, void* d_ws, size_t ws_size,
                              hipStream_t stream) {
  (void)in_sizes; (void)n_in; (void)out_size; (void)ws_size;
  const float* x   = (const float*)d_in[0];
  const float* W[3]  = {(const float*)d_in[1], (const float*)d_in[4], (const float*)d_in[7]};
  const float* U[3]  = {(const float*)d_in[2], (const float*)d_in[5], (const float*)d_in[8]};
  const float* bL[3] = {(const float*)d_in[3], (const float*)d_in[6], (const float*)d_in[9]};
  const float* bn_g[3] = {(const float*)d_in[10], (const float*)d_in[14], (const float*)d_in[18]};
  const float* bn_b[3] = {(const float*)d_in[11], (const float*)d_in[15], (const float*)d_in[19]};
  const float* bn_m[3] = {(const float*)d_in[12], (const float*)d_in[16], (const float*)d_in[20]};
  const float* bn_v[3] = {(const float*)d_in[13], (const float*)d_in[17], (const float*)d_in[21]};
  const float* Wa  = (const float*)d_in[22];  const float* ba  = (const float*)d_in[23];
  const float* Wd1 = (const float*)d_in[24];  const float* bd1 = (const float*)d_in[25];
  const float* Wd2 = (const float*)d_in[26];  const float* bd2 = (const float*)d_in[27];
  const float* Wd3 = (const float*)d_in[28];  const float* bd3 = (const float*)d_in[29];

  // workspace layout (all offsets naturally 256B-aligned)
  char* ws = (char*)d_ws;
  size_t o = 0;
  float* XZ = (float*)(ws + o);              o += (size_t)MM * FH * 4;   // 512 MB (reused as e)
  unsigned short* HSEQB = (unsigned short*)(ws + o); o += (size_t)MM * HH * 2; // 64 MB
  float* HSEQF = (float*)(ws + o);           o += (size_t)MM * HH * 4;   // 128 MB
  unsigned short* XBF = (unsigned short*)(ws + o);   o += (size_t)MM * DD * 2; // 16 MB
  unsigned short* WT[3]; unsigned short* UT[3];
  WT[0] = (unsigned short*)(ws + o); o += (size_t)FH * DD * 2;
  UT[0] = (unsigned short*)(ws + o); o += (size_t)FH * HH * 2;
  WT[1] = (unsigned short*)(ws + o); o += (size_t)FH * HH * 2;
  UT[1] = (unsigned short*)(ws + o); o += (size_t)FH * HH * 2;
  WT[2] = (unsigned short*)(ws + o); o += (size_t)FH * HH * 2;
  UT[2] = (unsigned short*)(ws + o); o += (size_t)FH * HH * 2;
  unsigned short* WAT = (unsigned short*)(ws + o); o += (size_t)HH * HH * 2;
  unsigned short* HST0 = (unsigned short*)(ws + o); o += (size_t)BB * HH * 2;
  unsigned short* HST1 = (unsigned short*)(ws + o); o += (size_t)BB * HH * 2;
  float* CST  = (float*)(ws + o); o += (size_t)BB * HH * 4;
  float* SBUF = (float*)(ws + o); o += (size_t)BB * TT * 4;
  float* ABUF = (float*)(ws + o); o += (size_t)BB * TT * 4;
  float* POOL = (float*)(ws + o); o += (size_t)BB * HH * 4;
  float* D1   = (float*)(ws + o); o += (size_t)BB * 128 * 4;
  float* D2   = (float*)(ws + o); o += (size_t)BB * 64 * 4;
  float* EBUF = XZ;   // attention scores reuse the xz region

  // weight prep: transpose to [N,K] bf16 so WMMA B-fragments are contiguous b128 loads
  transpose_bf16<<<(FH * DD + 255) / 256, 256, 0, stream>>>(WT[0], W[0], DD, FH);
  transpose_bf16<<<(FH * HH + 255) / 256, 256, 0, stream>>>(UT[0], U[0], HH, FH);
  transpose_bf16<<<(FH * HH + 255) / 256, 256, 0, stream>>>(WT[1], W[1], HH, FH);
  transpose_bf16<<<(FH * HH + 255) / 256, 256, 0, stream>>>(UT[1], U[1], HH, FH);
  transpose_bf16<<<(FH * HH + 255) / 256, 256, 0, stream>>>(WT[2], W[2], HH, FH);
  transpose_bf16<<<(FH * HH + 255) / 256, 256, 0, stream>>>(UT[2], U[2], HH, FH);
  transpose_bf16<<<(HH * HH + 255) / 256, 256, 0, stream>>>(WAT, Wa, HH, HH);
  cast_bf16<<<(MM * DD + 255) / 256, 256, 0, stream>>>(XBF, x, MM * DD);

  for (int layer = 0; layer < 3; ++layer) {
    // input projection (time-parallel WMMA GEMM): xz = in @ W + b
    if (layer == 0)
      gemm_bf16<<<dim3(FH / 64, MM / 128), 256, 0, stream>>>(XZ, XBF, WT[0], bL[0], MM, FH, DD);
    else
      gemm_bf16<<<dim3(FH / 64, MM / 128), 256, 0, stream>>>(XZ, HSEQB, WT[layer], bL[layer], MM, FH, HH);

    // zero h (both buffers) and c state
    fill_u32<<<(BB * HH / 2 + 255) / 256, 256, 0, stream>>>((unsigned int*)HST0, 0u, BB * HH / 2);
    fill_u32<<<(BB * HH / 2 + 255) / 256, 256, 0, stream>>>((unsigned int*)HST1, 0u, BB * HH / 2);
    fill_u32<<<(BB * HH + 255) / 256, 256, 0, stream>>>((unsigned int*)CST, 0u, BB * HH);

    const int last = (layer == 2);
    unsigned short* hr = HST0;
    unsigned short* hw = HST1;
    for (int t = 0; t < TT; ++t) {
      lstm_step<<<dim3(HH / 16), 256, 0, stream>>>(
          XZ, UT[layer], hr, hw, CST, HSEQB, last ? HSEQF : nullptr,
          last ? nullptr : bn_g[layer], last ? nullptr : bn_b[layer],
          last ? nullptr : bn_m[layer], last ? nullptr : bn_v[layer], t);
      unsigned short* tmp = hr; hr = hw; hw = tmp;
    }
  }

  // attention: e = hseq2 @ Wa + ba ; s = sum_k tanh(e) ; a = softmax_t(s)
  gemm_bf16<<<dim3(HH / 64, MM / 128), 256, 0, stream>>>(EBUF, HSEQB, WAT, ba, MM, HH, HH);
  attn_score<<<MM / 8, 256, 0, stream>>>(SBUF, EBUF);
  softmax_t<<<BB, 128, 0, stream>>>(ABUF, SBUF);
  weighted_mean<<<(BB * HH) / 256, 256, 0, stream>>>(POOL, HSEQF, ABUF);

  // dense head (tiny): d1 = BN2(relu(pool@Wd1+b)); d2 = relu(d1@Wd2+b); out = d2@Wd3+b
  dense_k<<<(BB * 128 + 255) / 256, 256, 0, stream>>>(D1, POOL, Wd1, bd1, BB, HH, 128, 1,
                                                      bn_g[2], bn_b[2], bn_m[2], bn_v[2]);
  dense_k<<<(BB * 64 + 255) / 256, 256, 0, stream>>>(D2, D1, Wd2, bd2, BB, 128, 64, 1,
                                                     nullptr, nullptr, nullptr, nullptr);
  dense_k<<<(BB * 5 + 255) / 256, 256, 0, stream>>>((float*)d_out, D2, Wd3, bd3, BB, 64, 5, 0,
                                                    nullptr, nullptr, nullptr, nullptr);
}
